// PostProcess_30116310679585
// MI455X (gfx1250) — matrix operand order repaired and checked by reading, not compiled, not run on hardware
//
#include <hip/hip_runtime.h>
#include <hip/hip_bf16.h>
#include <math.h>

// ---------------------------------------------------------------------------
// Types for CDNA5 WMMA (wave32): A/B = v16h (16 halfs = 8 VGPRs), C/D = v8f.
// ---------------------------------------------------------------------------
typedef __attribute__((ext_vector_type(16))) _Float16 v16h;
typedef __attribute__((ext_vector_type(8)))  _Float16 v8h;
typedef __attribute__((ext_vector_type(8)))  float    v8f;
typedef __attribute__((ext_vector_type(4)))  float    v4f;

#define GT_L   128           // mask side
#define NKEEP  256           // kept zigzag coefficients
#define NB     8
#define NR     300
#define NC     91
#define NITEMS (NB * NR)     // 2400

// ---------------------------------------------------------------------------
// WMMA fragment loaders from LDS, 16-bit data, K = 32.
//
// A-matrix 16x32 (MxK), contiguous k order used by this code:
//   lanes 0-15  (M = lane):     v[0..15] = K 0..15
//   lanes 16-31 (M = lane-16):  v[0..15] = K 16..31
// Source stored row-major [row][k], row stride 32 halfs (64 B) -> two
// contiguous 16-byte chunks per lane => two ds_load_b128.
//
// B-matrix 32x16 (KxN), same contiguous k order as A:
//   lanes 0-15  (N = lane):     K 0..15 packed 2/VGPR
//   lanes 16-31 (N = lane-16):  K 16..31
// Source stored transposed [n][k], row stride 32 halfs -> one contiguous
// 32-byte run per lane => two ds_load_b128.
// ---------------------------------------------------------------------------
__device__ inline v16h load_fragA(const _Float16* base, int row, int laneHi) {
    const _Float16* p = base + row * 32;
    union { v16h v; v8h h[2]; } u;
    u.h[0] = *(const v8h*)(p + (laneHi ? 16 : 0));
    u.h[1] = *(const v8h*)(p + 8 + (laneHi ? 16 : 0));
    return u.v;
}

__device__ inline v16h load_fragB(const _Float16* base, int row, int laneHi) {
    const _Float16* p = base + row * 32 + (laneHi ? 16 : 0);
    union { v16h v; v8h h[2]; } u;
    u.h[0] = *(const v8h*)(p);
    u.h[1] = *(const v8h*)(p + 8);
    return u.v;
}

// ---------------------------------------------------------------------------
// Main kernel: one workgroup (8 wave32) per (b, r) item.
//   M = C^T (128x23) * K (23x23) * C (23x128); K zigzag-scattered, zero-pad
//   K-dim 23 -> 32 for 16x16x32 f16 WMMA; binarize at (max+min)/2; stream
//   the 128x128 f32 mask to HBM with fully coalesced b128 row stores.
//
// LDS: the f16 staging arrays (18 KB) are dead once GEMM2 fragments are in
// registers, so they are overlaid (union) with the 64 KB f32 store-staging
// buffer used for coalescing the output.
// ---------------------------------------------------------------------------
__global__ __launch_bounds__(256) void masks_kernel(const float* __restrict__ pv,
                                                    float* __restrict__ out_masks) {
    __shared__ union {
        struct {
            __align__(16) _Float16 coefA[32 * 32];   // K zigzag tile   (2 KB)
            __align__(16) _Float16 CT[GT_L * 32];    // CT[m][k]=C[k][m] (8 KB)
            __align__(16) _Float16 T1T[GT_L * 32];   // T1T[n][k]=T1[k][n] (8 KB)
        } s;
        __align__(16) float stage[GT_L * GT_L];      // mask staging (64 KB)
    } u;
    __shared__ float redmax[8];
    __shared__ float redmin[8];

    const int item = blockIdx.x;
    const int tid  = threadIdx.x;
    const int lane = tid & 31;
    const int wave = tid >> 5;
    const int lhi  = (lane >= 16) ? 1 : 0;
    const int lmod = lane & 15;

    // ---- stage 0: zero K tile, build CT[m][k] = C[k][m] -----------------
    for (int idx = tid; idx < 32 * 32; idx += 256) u.s.coefA[idx] = (_Float16)0.0f;

    const float SQ1 = 0.08838834764831845f;  // sqrt(1/128)
    const float SQ2 = 0.125f;                // sqrt(2/128)
    const float PIC = 3.14159265358979323846f / 256.0f;
    for (int idx = tid; idx < GT_L * 32; idx += 256) {
        const int m = idx >> 5;
        const int k = idx & 31;
        float v = (k == 0) ? SQ1 : SQ2 * __cosf(PIC * (float)((2 * m + 1) * k));
        u.s.CT[idx] = (_Float16)v;
    }
    __syncthreads();

    // ---- stage 1: zigzag scatter of 256 coefficients (all land i+j<=22) -
    {
        const int k = tid;                 // 256 threads <-> 256 coefficients
        int s = 0, base = 0;
        while (base + s + 1 <= k) { base += s + 1; ++s; }
        const int t = k - base;
        const int i = (s & 1) ? t : (s - t);
        const int j = (s & 1) ? (s - t) : t;
        u.s.coefA[i * 32 + j] = (_Float16)pv[(size_t)item * NKEEP + k];
    }
    __syncthreads();

    // ---- stage 2: GEMM1  T1(32x128) = K(32x32) @ C(32x128) --------------
    // 16 output 16x16 tiles; each wave does 2 (statically unrolled).
    // Result stored transposed into T1T[n][k] for contiguous GEMM2 B loads.
#pragma unroll
    for (int tt = 0; tt < 2; ++tt) {
        const int t  = wave + (tt << 3);
        const int i0 = (t >> 3) << 4;      // 0 or 16
        const int n0 = (t & 7) << 4;       // 0..112
        v16h a = load_fragA(u.s.coefA, i0 + lmod, lhi);  // K rows
        v16h b = load_fragB(u.s.CT,    n0 + lmod, lhi);  // B[k][n] = CT[n][k]
        v8f  c = {};
        c = __builtin_amdgcn_wmma_f32_16x16x32_f16(false, a, false, b,
                                                   (short)0, c, false, false);
        const int n  = n0 + lmod;
        const int ib = i0 + (lhi ? 8 : 0);
#pragma unroll
        for (int g = 0; g < 8; ++g)
            u.s.T1T[n * 32 + ib + g] = (_Float16)c[g];
    }
    __syncthreads();

    // ---- stage 3: GEMM2  M(128x128) = C^T(128x32) @ T1(32x128) ----------
    // Wave w owns rows m0..m0+15; 8 accumulator tiles (64 VGPRs) stay live.
    const int m0 = wave << 4;
    const v16h a2 = load_fragA(u.s.CT, m0 + lmod, lhi);   // A2[m][k] = CT[m][k]
    v8f acc[8];
#pragma unroll
    for (int t = 0; t < 8; ++t) {
        v16h b2 = load_fragB(u.s.T1T, (t << 4) + lmod, lhi); // B2[k][n] = T1T[n][k]
        v8f  c  = {};
        acc[t] = __builtin_amdgcn_wmma_f32_16x16x32_f16(false, a2, false, b2,
                                                        (short)0, c, false, false);
    }

    // ---- stage 4: max/min reduction over all 128x128 values -------------
    float mx = acc[0][0], mn = acc[0][0];
#pragma unroll
    for (int t = 0; t < 8; ++t)
#pragma unroll
        for (int g = 0; g < 8; ++g) {
            const float v = acc[t][g];
            mx = fmaxf(mx, v);
            mn = fminf(mn, v);
        }
#pragma unroll
    for (int off = 16; off >= 1; off >>= 1) {
        mx = fmaxf(mx, __shfl_xor(mx, off, 32));
        mn = fminf(mn, __shfl_xor(mn, off, 32));
    }
    if (lane == 0) { redmax[wave] = mx; redmin[wave] = mn; }
    __syncthreads();   // also fences all T1T/CT ds reads before union reuse
    float gmx = redmax[0], gmn = redmin[0];
#pragma unroll
    for (int w = 1; w < 8; ++w) {
        gmx = fmaxf(gmx, redmax[w]);
        gmn = fminf(gmn, redmin[w]);
    }
    const float thr = (gmx + gmn) * 0.5f;

    // ---- stage 5: threshold into LDS (D-tile layout), then stream out ---
    // Wave-private 16x128 strip: LDS store->load is same-wave & in-order,
    // so no workgroup barrier is needed.
    {
        const int mrow = m0 + (lhi ? 8 : 0);
#pragma unroll
        for (int t = 0; t < 8; ++t) {
            const int n = (t << 4) + lmod;
#pragma unroll
            for (int g = 0; g < 8; ++g)
                u.stage[(mrow + g) * GT_L + n] = (acc[t][g] > thr) ? 1.0f : 0.0f;
        }
    }
    // Coalesced write-out: iteration i => one full 512-B mask row per wave
    // (32 lanes x 16 B global_store_b128).
    float* __restrict__ outp = out_masks + (size_t)item * (GT_L * GT_L);
#pragma unroll
    for (int i = 0; i < 16; ++i) {
        const int row  = m0 + i;
        const int col4 = lane << 2;                      // float4 column
        v4f vv = *(const v4f*)&u.stage[row * GT_L + col4];
        *(v4f*)&outp[(size_t)row * GT_L + col4] = vv;
    }
}

// ---------------------------------------------------------------------------
// Small head kernel: scores (max sigmoid), labels (argmax, first-wins),
// scaled boxes. One thread per (b, r).
// ---------------------------------------------------------------------------
__global__ void head_kernel(const float* __restrict__ logits,
                            const float* __restrict__ boxes,
                            const int*   __restrict__ tsizes,
                            float* __restrict__ o_scores,
                            float* __restrict__ o_labels,
                            float* __restrict__ o_boxes) {
    const int idx = blockIdx.x * blockDim.x + threadIdx.x;
    if (idx >= NITEMS) return;
    const int b = idx / NR;

    const float* lg = logits + (size_t)idx * NC;
    float best = lg[0];
    int   bl   = 0;
    for (int c = 1; c < NC; ++c) {
        const float v = lg[c];
        if (v > best) { best = v; bl = c; }   // strict > keeps first max (jnp.argmax)
    }
    o_scores[idx] = 1.0f / (1.0f + __expf(-best));   // sigmoid of max logit
    o_labels[idx] = (float)bl;

    const float cx = boxes[idx * 4 + 0];
    const float cy = boxes[idx * 4 + 1];
    const float w  = boxes[idx * 4 + 2];
    const float h  = boxes[idx * 4 + 3];
    const float ih = (float)tsizes[b * 2 + 0];
    const float iw = (float)tsizes[b * 2 + 1];
    o_boxes[idx * 4 + 0] = (cx - 0.5f * w) * iw;
    o_boxes[idx * 4 + 1] = (cy - 0.5f * h) * ih;
    o_boxes[idx * 4 + 2] = (cx + 0.5f * w) * iw;
    o_boxes[idx * 4 + 3] = (cy + 0.5f * h) * ih;
}

// ---------------------------------------------------------------------------
// d_out layout (reference return order, flat):
//   scores[2400] | labels[2400] | boxes[2400*4] | masks[2400*128*128]
// ---------------------------------------------------------------------------
extern "C" void kernel_launch(void* const* d_in, const int* in_sizes, int n_in,
                              void* d_out, int out_size, void* d_ws, size_t ws_size,
                              hipStream_t stream) {
    const float* logits = (const float*)d_in[0];   // (8,300,91)  f32
    const float* boxes  = (const float*)d_in[1];   // (8,300,4)   f32
    const float* pv     = (const float*)d_in[2];   // (8,300,256) f32
    const int*   ts     = (const int*)  d_in[3];   // (8,2)       i32

    float* out      = (float*)d_out;
    float* o_scores = out;
    float* o_labels = out + NITEMS;
    float* o_boxes  = out + 2 * NITEMS;
    float* o_masks  = out + 2 * NITEMS + 4 * NITEMS;

    head_kernel<<<(NITEMS + 255) / 256, 256, 0, stream>>>(logits, boxes, ts,
                                                          o_scores, o_labels, o_boxes);
    masks_kernel<<<NITEMS, 256, 0, stream>>>(pv, o_masks);
}
